// LSPConditionalGNN_59236188947122
// MI455X (gfx1250) — matrix operand order repaired and checked by reading, not compiled, not run on hardware
//
#include <hip/hip_runtime.h>
#include <hip/hip_bf16.h>

// ---------------------------------------------------------------------------
// 3-layer GCN (PyG GCNConv semantics) for MI455X / gfx1250.
//  - GEMMs: v_wmma_f32_16x16x32_bf16, A pre-tiled to fragment order
//    (coalesced b128 loads), B k-slices double-buffered through LDS.
//  - Scatter: precomputed edge norms + global_atomic_add_f32 (HBM-bound).
// ---------------------------------------------------------------------------

typedef __attribute__((ext_vector_type(16))) __bf16 v16bf;
typedef __attribute__((ext_vector_type(8)))  float  v8f;

union FragBF { v16bf v; uint4 u[2]; };

static constexpr int NN  = 200000;   // nodes (N/16 = 12500 strips = 3125 blocks * 4 waves)
static constexpr int NE  = 3200000;  // edges
static constexpr int KP1 = 288;      // 258 padded to multiple of 32
static constexpr int F1  = 128;
static constexpr int F2  = 64;
static constexpr int F3  = 8;
static constexpr int NP3 = 16;       // F3 padded to 16 for WMMA

// ---------------------------------------------------------------------------
// degree / norm
// ---------------------------------------------------------------------------
__global__ void k_init_deg(float* deg, int n) {
    int i = blockIdx.x * blockDim.x + threadIdx.x;
    if (i < n) deg[i] = 1.0f;  // self loop
}

__global__ void k_edge_deg(const int* __restrict__ dst, float* deg, int e) {
    int i = blockIdx.x * blockDim.x + threadIdx.x;
    if (i < e) atomicAdd(&deg[dst[i]], 1.0f);
}

__global__ void k_dinv(float* deg, int n) {
    int i = blockIdx.x * blockDim.x + threadIdx.x;
    if (i < n) {
        float d = deg[i];
        deg[i] = d > 0.0f ? rsqrtf(d) : 0.0f;  // in-place: deg -> dinv
    }
}

__global__ void k_edge_norm(const int* __restrict__ src, const int* __restrict__ dst,
                            const float* __restrict__ dinv, float* __restrict__ norm, int e) {
    int i = blockIdx.x * blockDim.x + threadIdx.x;
    if (i < e) norm[i] = dinv[src[i]] * dinv[dst[i]];
}

// ---------------------------------------------------------------------------
// A-matrix fragment tiling helpers.
// Apack index((strip,kt,lane,e)) = ((strip*KT + kt)*32 + lane)*16 + e
//   row = strip*16 + (lane&15), g = lane>>4
//   k   = kt*32 + (e<8 ? g*8+e : 16 + g*8 + (e-8))     (ISA 16-bit A 16x32 layout)
// ---------------------------------------------------------------------------
__device__ __forceinline__ void a_decompose(long long t, int KT, int& row, int& k) {
    int e    = (int)(t & 15);
    int lane = (int)((t >> 4) & 31);
    long long rest = t >> 9;
    int kt = (int)(rest % KT);
    int s  = (int)(rest / KT);
    int g  = lane >> 4;
    row = s * 16 + (lane & 15);
    int ke = (e < 8) ? (g * 8 + e) : (16 + g * 8 + (e - 8));
    k = kt * 32 + ke;
}

// pack [latent|history|subgoal] (fp32) -> Apack bf16, K padded 258->288
__global__ void k_pack_features(const float* __restrict__ lat,
                                const float* __restrict__ hist,
                                const float* __restrict__ sub,
                                __bf16* __restrict__ h, long long total) {
    long long t = (long long)blockIdx.x * blockDim.x + threadIdx.x;
    if (t >= total) return;
    int row, k;
    a_decompose(t, KP1 / 32, row, k);
    float v;
    if (k < 256)       v = lat[(long long)row * 256 + k];
    else if (k == 256) v = hist[row];
    else if (k == 257) v = sub[row];
    else               v = 0.0f;
    h[t] = (__bf16)v;
}

// relu(agg + b) (fp32) -> Apack bf16 for next layer (F == KPADN here)
template <int F, int KPADN>
__global__ void k_bias_relu_pack(const float* __restrict__ agg, const float* __restrict__ b,
                                 __bf16* __restrict__ h, long long total) {
    long long t = (long long)blockIdx.x * blockDim.x + threadIdx.x;
    if (t >= total) return;
    int row, k;
    a_decompose(t, KPADN / 32, row, k);
    float v = 0.0f;
    if (k < F) {
        v = agg[(long long)row * F + k] + b[k];
        v = v > 0.0f ? v : 0.0f;
    }
    h[t] = (__bf16)v;
}

// Pre-swizzle weight matrix W[KIN x NIN] (row-major f32) into B fragments:
// Bp[((kt*NT + nt)*32 + lane)*16 + e] covers K = kt*32 + (e<8 ? (lane>>4)*8+e : 16+(lane>>4)*8+(e-8)),
// N = nt*16 + (lane&15).  Zero-pad beyond KIN/NIN.
template <int KIN, int NIN, int KPAD, int NPAD>
__global__ void k_pack_W(const float* __restrict__ W, __bf16* __restrict__ Bp) {
    constexpr int NT = NPAD / 16;
    int t = blockIdx.x * blockDim.x + threadIdx.x;
    if (t >= KPAD * NPAD) return;
    int e    = t & 15;
    int lane = (t >> 4) & 31;
    int rest = t >> 9;
    int nt = rest % NT;
    int kt = rest / NT;
    int k = kt * 32 + ((e < 8) ? ((lane >> 4) * 8 + e) : (16 + (lane >> 4) * 8 + (e - 8)));
    int n = nt * 16 + (lane & 15);
    float v = (k < KIN && n < NIN) ? W[k * NIN + n] : 0.0f;
    Bp[t] = (__bf16)v;
}

// ---------------------------------------------------------------------------
// WMMA GEMM: C[n x NOUT] = Apack(bf16) * Bp(pre-swizzled bf16)
// 128-thread blocks = 4 waves, one 16-row strip per wave (exact cover).
// B k-slices double-buffered through LDS.  Staging uses power-of-two masked
// indices (no divergence, no scratch spill of the staging registers); small
// slices are written redundantly with identical values (benign).
// ---------------------------------------------------------------------------
template <int KPAD, int NPAD, int NOUT>
__global__ void __launch_bounds__(128)
k_gemm_bf16(const __bf16* __restrict__ A, const __bf16* __restrict__ Bp,
            float* __restrict__ C, int strips) {
    constexpr int NT     = NPAD / 16;
    constexpr int KT     = KPAD / 32;
    constexpr int SLICE4 = NT * 64;                 // uint4 per k-slice of B (pow2)
    constexpr int MASK   = SLICE4 - 1;
    constexpr int NSTG   = (SLICE4 + 127) / 128;    // staged uint4 per thread
    __shared__ uint4 sB[2][SLICE4];

    const int tid   = threadIdx.x;
    const int lane  = tid & 31;
    const int g     = lane >> 4;
    const int strip = blockIdx.x * 4 + (tid >> 5);
    if (strip >= strips) return;
    const int m0 = strip * 16;

    const uint4* gB = reinterpret_cast<const uint4*>(Bp);
    const uint4* pA = reinterpret_cast<const uint4*>(A) +
                      ((long long)strip * KT * 32 + lane) * 2;

    // stage k-slice 0 (unconditional, masked index)
#pragma unroll
    for (int j = 0; j < NSTG; ++j) {
        int idx = (tid + j * 128) & MASK;
        sB[0][idx] = gB[idx];
    }

    v8f acc[NT];
    v8f z = {0.f, 0.f, 0.f, 0.f, 0.f, 0.f, 0.f, 0.f};
#pragma unroll
    for (int nt = 0; nt < NT; ++nt) acc[nt] = z;

    FragBF a_cur, a_nxt;
    a_cur.u[0] = pA[0];
    a_cur.u[1] = pA[1];
    __syncthreads();

#pragma unroll
    for (int kt = 0; kt < KT; ++kt) {
        const int buf = kt & 1;
        uint4 stg[NSTG];
        if (kt + 1 < KT) {                       // compile-time resolvable (unrolled)
            const uint4* gBn = gB + (long long)(kt + 1) * SLICE4;
#pragma unroll
            for (int j = 0; j < NSTG; ++j)
                stg[j] = gBn[(tid + j * 128) & MASK];
            a_nxt.u[0] = pA[(kt + 1) * 64];
            a_nxt.u[1] = pA[(kt + 1) * 64 + 1];
        }

        // compute against current LDS slice (overlaps the loads above)
#pragma unroll
        for (int nt = 0; nt < NT; ++nt) {
            const uint4* sb = &sB[buf][(nt * 32 + lane) * 2];
            FragBF b;
            b.u[0] = sb[0];
            b.u[1] = sb[1];
            acc[nt] = __builtin_amdgcn_wmma_f32_16x16x32_bf16(
                false, a_cur.v, false, b.v, (short)0, acc[nt], false, false);
        }

        if (kt + 1 < KT) {
            // drain staged B into the other LDS buffer
#pragma unroll
            for (int j = 0; j < NSTG; ++j)
                sB[buf ^ 1][(tid + j * 128) & MASK] = stg[j];
            a_cur = a_nxt;
        }
        __syncthreads();
    }

    // store C (two 64B coalesced segments per instruction)
#pragma unroll
    for (int nt = 0; nt < NT; ++nt) {
        int n = nt * 16 + (lane & 15);
        if (n < NOUT) {
#pragma unroll
            for (int r = 0; r < 8; ++r)
                C[(long long)(m0 + r + 8 * g) * NOUT + n] = acc[nt][r];
        }
    }
}

// ---------------------------------------------------------------------------
// scatter: agg = self-loop init, then edge atomics
// ---------------------------------------------------------------------------
template <int F>
__global__ void k_self_loop(const float* __restrict__ tmp, const float* __restrict__ dinv,
                            float* __restrict__ agg, int n) {
    constexpr int C4 = F / 4;
    long long t = (long long)blockIdx.x * blockDim.x + threadIdx.x;
    if (t >= (long long)n * C4) return;
    int i  = (int)(t / C4);
    int c4 = (int)(t % C4) * 4;
    float s = dinv[i];
    s = s * s;
    const float4 v = *reinterpret_cast<const float4*>(tmp + (long long)i * F + c4);
    float4 o = make_float4(s * v.x, s * v.y, s * v.z, s * v.w);
    *reinterpret_cast<float4*>(agg + (long long)i * F + c4) = o;
}

template <int F>
__global__ void k_edge_scatter(const int* __restrict__ src, const int* __restrict__ dst,
                               const float* __restrict__ norm,
                               const float* __restrict__ tmp, float* __restrict__ agg,
                               int e) {
    constexpr int C4 = F / 4;
    long long t = (long long)blockIdx.x * blockDim.x + threadIdx.x;
    if (t >= (long long)e * C4) return;
    int ei = (int)(t / C4);
    int c4 = (int)(t % C4) * 4;
    float w = norm[ei];
    long long so  = (long long)src[ei] * F + c4;
    long long dof = (long long)dst[ei] * F + c4;
    const float4 v = *reinterpret_cast<const float4*>(tmp + so);
    atomicAdd(&agg[dof + 0], w * v.x);
    atomicAdd(&agg[dof + 1], w * v.y);
    atomicAdd(&agg[dof + 2], w * v.z);
    atomicAdd(&agg[dof + 3], w * v.w);
}

// final: relu(agg3 + b3) @ Wc + bc  -> out [N x 3]
__global__ void k_head(const float* __restrict__ agg, const float* __restrict__ b3,
                       const float* __restrict__ Wc, const float* __restrict__ bc,
                       float* __restrict__ out, int n) {
    int i = blockIdx.x * blockDim.x + threadIdx.x;
    if (i >= n) return;
    float h[8];
#pragma unroll
    for (int k = 0; k < 8; ++k) {
        float v = agg[(long long)i * 8 + k] + b3[k];
        h[k] = v > 0.0f ? v : 0.0f;
    }
#pragma unroll
    for (int j = 0; j < 3; ++j) {
        float s = bc[j];
#pragma unroll
        for (int k = 0; k < 8; ++k) s += h[k] * Wc[k * 3 + j];
        out[(long long)i * 3 + j] = s;
    }
}

// ---------------------------------------------------------------------------
// host
// ---------------------------------------------------------------------------
static inline unsigned cdiv(long long a, long long b) { return (unsigned)((a + b - 1) / b); }

extern "C" void kernel_launch(void* const* d_in, const int* in_sizes, int n_in,
                              void* d_out, int out_size, void* d_ws, size_t ws_size,
                              hipStream_t stream) {
    (void)in_sizes; (void)n_in; (void)out_size; (void)ws_size;
    const float* lat  = (const float*)d_in[0];
    const float* hist = (const float*)d_in[1];
    const float* sub  = (const float*)d_in[2];
    const int*   ei   = (const int*)d_in[3];   // [2, E] int32
    const float* W1   = (const float*)d_in[4];
    const float* b1   = (const float*)d_in[5];
    const float* W2   = (const float*)d_in[6];
    const float* b2   = (const float*)d_in[7];
    const float* W3   = (const float*)d_in[8];
    const float* b3   = (const float*)d_in[9];
    const float* Wc   = (const float*)d_in[10];
    const float* bc   = (const float*)d_in[11];
    float* out = (float*)d_out;

    const int* src = ei;
    const int* dst = ei + NE;

    // workspace layout (256B aligned slices)
    char* ws = (char*)d_ws;
    size_t off = 0;
    auto alloc = [&](size_t bytes) { size_t r = off; off = (off + bytes + 255) & ~(size_t)255; return r; };
    float*  dinv = (float*) (ws + alloc((size_t)NN * 4));
    float*  norm = (float*) (ws + alloc((size_t)NE * 4));
    __bf16* hbuf = (__bf16*)(ws + alloc((size_t)NN * KP1 * 2));  // Apack, reused each layer
    float*  tmp  = (float*) (ws + alloc((size_t)NN * F1 * 4));   // GEMM output, reused
    float*  agg  = (float*) (ws + alloc((size_t)NN * F1 * 4));   // scatter accumulator, reused
    __bf16* bp1  = (__bf16*)(ws + alloc((size_t)(KP1 / 32) * (F1 / 16) * 512 * 2));
    __bf16* bp2  = (__bf16*)(ws + alloc((size_t)(F1 / 32) * (F2 / 16) * 512 * 2));
    __bf16* bp3  = (__bf16*)(ws + alloc((size_t)(F2 / 32) * (NP3 / 16) * 512 * 2));

    const int T = 256;
    const int strips = NN / 16;                 // 12500
    const unsigned gemm_blocks = strips / 4;    // 3125 blocks * 4 waves, exact

    // degree / normalization
    k_init_deg<<<cdiv(NN, T), T, 0, stream>>>(dinv, NN);
    k_edge_deg<<<cdiv(NE, T), T, 0, stream>>>(dst, dinv, NE);
    k_dinv<<<cdiv(NN, T), T, 0, stream>>>(dinv, NN);
    k_edge_norm<<<cdiv(NE, T), T, 0, stream>>>(src, dst, dinv, norm, NE);

    // pack inputs + weights
    k_pack_features<<<cdiv((long long)NN * KP1, T), T, 0, stream>>>(lat, hist, sub, hbuf, (long long)NN * KP1);
    k_pack_W<258, F1, KP1, F1><<<cdiv(KP1 * F1, T), T, 0, stream>>>(W1, bp1);
    k_pack_W<F1, F2, F1, F2><<<cdiv(F1 * F2, T), T, 0, stream>>>(W2, bp2);
    k_pack_W<F2, F3, F2, NP3><<<cdiv(F2 * NP3, T), T, 0, stream>>>(W3, bp3);

    // ---- layer 1: 288 -> 128
    k_gemm_bf16<KP1, F1, F1><<<gemm_blocks, 128, 0, stream>>>(hbuf, bp1, tmp, strips);
    k_self_loop<F1><<<cdiv((long long)NN * (F1 / 4), T), T, 0, stream>>>(tmp, dinv, agg, NN);
    k_edge_scatter<F1><<<cdiv((long long)NE * (F1 / 4), T), T, 0, stream>>>(src, dst, norm, tmp, agg, NE);
    k_bias_relu_pack<F1, F1><<<cdiv((long long)NN * F1, T), T, 0, stream>>>(agg, b1, hbuf, (long long)NN * F1);

    // ---- layer 2: 128 -> 64
    k_gemm_bf16<F1, F2, F2><<<gemm_blocks, 128, 0, stream>>>(hbuf, bp2, tmp, strips);
    k_self_loop<F2><<<cdiv((long long)NN * (F2 / 4), T), T, 0, stream>>>(tmp, dinv, agg, NN);
    k_edge_scatter<F2><<<cdiv((long long)NE * (F2 / 4), T), T, 0, stream>>>(src, dst, norm, tmp, agg, NE);
    k_bias_relu_pack<F2, F2><<<cdiv((long long)NN * F2, T), T, 0, stream>>>(agg, b2, hbuf, (long long)NN * F2);

    // ---- layer 3: 64 -> 8 (padded to 16 for WMMA, compact f32 output)
    k_gemm_bf16<F2, NP3, F3><<<gemm_blocks, 128, 0, stream>>>(hbuf, bp3, tmp, strips);
    k_self_loop<F3><<<cdiv((long long)NN * (F3 / 4), T), T, 0, stream>>>(tmp, dinv, agg, NN);
    k_edge_scatter<F3><<<cdiv((long long)NE * (F3 / 4), T), T, 0, stream>>>(src, dst, norm, tmp, agg, NE);

    // ---- head: relu(agg + b3) @ Wc + bc
    k_head<<<cdiv(NN, T), T, 0, stream>>>(agg, b3, Wc, bc, out, NN);
}
